// MILPoolingLayer_69844758168190
// MI455X (gfx1250) — compile-verified, matrix operand order checked by reading, not run on hardware
//
#include <hip/hip_runtime.h>
#include <stdint.h>

// Problem constants (from reference): B=32, N=4096, M=256, L=128, H=4, GAMMA=2
#define B_ 32
#define N_ 4096
#define M_ 256
#define L_ 128
#define H_ 4

typedef __attribute__((ext_vector_type(16))) __bf16 v16bf;
typedef __attribute__((ext_vector_type(8)))  float  v8f;
typedef __attribute__((ext_vector_type(4)))  unsigned int v4u;
typedef __attribute__((ext_vector_type(4)))  int v4i;

union Frag16 { v4u q[2]; v16bf v; };

__device__ __forceinline__ uint16_t f2bf(float f) {
  uint32_t u = __float_as_uint(f);
  u += 0x7FFFu + ((u >> 16) & 1u);     // round-to-nearest-even (ignoring NaN edge)
  return (uint16_t)(u >> 16);
}

// ---------------- gfx1250 async global->LDS path ----------------
#if __has_builtin(__builtin_amdgcn_global_load_async_to_lds_b128)
#define HAVE_ASYNC_LDS 1
#endif

__device__ __forceinline__ void copy16_g2lds(const uint16_t* g, uint16_t* l) {
#ifdef HAVE_ASYNC_LDS
  typedef __attribute__((address_space(1))) v4i gvec_t;
  typedef __attribute__((address_space(3))) v4i lvec_t;
  __builtin_amdgcn_global_load_async_to_lds_b128(
      (gvec_t*)(unsigned long long)(uintptr_t)g,
      (lvec_t*)(unsigned int)(uintptr_t)l, 0, 0);
#else
  *(v4u*)l = *(const v4u*)g;
#endif
}

__device__ __forceinline__ void wait_async_lds() {
#ifdef HAVE_ASYNC_LDS
#if __has_builtin(__builtin_amdgcn_s_wait_asynccnt)
  __builtin_amdgcn_s_wait_asynccnt(0);
#else
  asm volatile("s_wait_asynccnt 0" ::: "memory");
#endif
#endif
}

// ================================================================
// Kernel 0: pack V,U (f32) into bf16 B-fragments (32x16 tiles) in d_ws.
// Fragment id F = ((Wm*H + h)*8 + lt)*8 + kt, 512 bf16 each, lane-major:
//   element (lane, e): N = lt*16 + (lane&15); hi = lane>>4; r=e>>1; c=e&1;
//   K = kt*32 + (r&3)*2 + c + ((r>>2)<<4) + (hi<<3)
// ================================================================
__global__ __launch_bounds__(256) void k_pack(const float* __restrict__ V,
                                              const float* __restrict__ U,
                                              uint16_t* __restrict__ wfrag) {
  int idx  = blockIdx.x * 256 + threadIdx.x;   // 0..262143
  int el   = idx & 15;
  int lane = (idx >> 4) & 31;
  int F    = idx >> 9;
  int kt = F & 7, lt = (F >> 3) & 7, h = (F >> 6) & 3, Wm = F >> 8;
  int r = el >> 1, c = el & 1, hi = lane >> 4;
  int K = kt * 32 + (r & 3) * 2 + c + ((r >> 2) << 4) + (hi << 3);
  int l = lt * 16 + (lane & 15);
  const float* src = Wm ? U : V;
  wfrag[idx] = f2bf(src[((long)h * M_ + K) * L_ + l]);
}

// ================================================================
// Kernel 1: fused  logits[b,h,n] = sum_l tanh(xV)*sigmoid(xU) * w[h,l]
// WG = 256 threads (8 waves), 128 rows of (B*N). 8 column groups of 64.
// ================================================================
__device__ __forceinline__ void stage_weights(const uint16_t* __restrict__ wfrag,
                                              uint16_t* dstLds, int cg, int t) {
  int h = cg >> 1, ltb = (cg & 1) * 4;
  const uint16_t* srcV = wfrag + (long)((h * 8 + ltb) * 8) * 512;
  const uint16_t* srcU = wfrag + (long)(((4 + h) * 8 + ltb) * 8) * 512;
#pragma unroll
  for (int i = 0; i < 8; ++i) {
    int off = (t + i * 256) * 8;               // 16B units over 32KB
    copy16_g2lds(srcV + off, dstLds + off);
    copy16_g2lds(srcU + off, dstLds + 16384 + off);
  }
}

__global__ __launch_bounds__(256) void k_gemm(const float* __restrict__ x,
                                              const uint16_t* __restrict__ wfrag,
                                              const float* __restrict__ wv,
                                              float* __restrict__ logits) {
  extern __shared__ char smem[];
  uint16_t* xf = (uint16_t*)smem;               // 8 rt * 8 kt * 512 = 64KB
  uint16_t* wb = (uint16_t*)(smem + 65536);     // 2 x 64KB double buffer

  const int t    = threadIdx.x;
  const int lane = t & 31;
  const int wave = t >> 5;                      // row tile 0..7
  const long rowBase = (long)blockIdx.x * 128;

  stage_weights(wfrag, wb, 0, t);               // async: first weight chunk

  // stage x tile (128 rows x 256 K) f32 -> bf16 A-fragment layout
#pragma unroll
  for (int i = 0; i < 32; ++i) {
    int idx = t + i * 256;                      // 0..8191 float4 chunks
    int row = idx >> 6;
    int K   = (idx & 63) * 4;
    const float4 v = *(const float4*)(x + (rowBase + row) * M_ + K);
    uint32_t lo = (uint32_t)f2bf(v.x) | ((uint32_t)f2bf(v.y) << 16);
    uint32_t hi2 = (uint32_t)f2bf(v.z) | ((uint32_t)f2bf(v.w) << 16);
    int kt = K >> 5;
    int hi = (K >> 3) & 1;
    int rb = ((K >> 4) & 1) * 4 + ((K >> 2) & 1) * 2;
    uint32_t* frag = (uint32_t*)(xf + ((row >> 4) * 8 + kt) * 512) +
                     ((row & 15) + 16 * hi) * 8 + rb;
    frag[0] = lo; frag[1] = hi2;
  }
  wait_async_lds();
  __syncthreads();

  float logitAcc[8];
#pragma unroll
  for (int r = 0; r < 8; ++r) logitAcc[r] = 0.f;

  for (int cg = 0; cg < 8; ++cg) {
    const int h = cg >> 1, lhalf = cg & 1;
    const uint16_t* wcur = wb + (cg & 1) * 32768;
    if (cg < 7) stage_weights(wfrag, wb + ((cg + 1) & 1) * 32768, cg + 1, t);

    v8f accV[4] = {};
    v8f accU[4] = {};

#pragma unroll
    for (int kt = 0; kt < 8; ++kt) {
      Frag16 a;
      const v4u* ap = (const v4u*)(xf + (wave * 8 + kt) * 512) + lane * 2;
      a.q[0] = ap[0]; a.q[1] = ap[1];
#pragma unroll
      for (int lt = 0; lt < 4; ++lt) {
        Frag16 bV, bU;
        const v4u* bp = (const v4u*)(wcur + (lt * 8 + kt) * 512) + lane * 2;
        bV.q[0] = bp[0]; bV.q[1] = bp[1];
        accV[lt] = __builtin_amdgcn_wmma_f32_16x16x32_bf16(
            false, a.v, false, bV.v, (short)0, accV[lt], false, false);
        const v4u* bq = (const v4u*)(wcur + 16384 + (lt * 8 + kt) * 512) + lane * 2;
        bU.q[0] = bq[0]; bU.q[1] = bq[1];
        accU[lt] = __builtin_amdgcn_wmma_f32_16x16x32_bf16(
            false, a.v, false, bU.v, (short)0, accU[lt], false, false);
      }
    }

    // epilogue: gate, dot with w[h,:], reduce across the 16 column lanes
    float rowPart[8];
#pragma unroll
    for (int r = 0; r < 8; ++r) rowPart[r] = 0.f;
#pragma unroll
    for (int lt = 0; lt < 4; ++lt) {
      const int l = lhalf * 64 + lt * 16 + (lane & 15);
      const float wl = wv[h * L_ + l];
#pragma unroll
      for (int r = 0; r < 8; ++r) {
        float e2 = __expf(2.0f * accV[lt][r]);
        float g  = (e2 - 1.0f) / (e2 + 1.0f);              // tanh
        float s  = 1.0f / (1.0f + __expf(-accU[lt][r]));   // sigmoid
        rowPart[r] += g * s * wl;
      }
    }
#pragma unroll
    for (int r = 0; r < 8; ++r) {
      float v = rowPart[r];
      v += __shfl_xor(v, 1); v += __shfl_xor(v, 2);
      v += __shfl_xor(v, 4); v += __shfl_xor(v, 8);
      logitAcc[r] += v;
    }
    if (lhalf == 1) {
      if ((lane & 15) == 0) {
#pragma unroll
        for (int r = 0; r < 8; ++r) {
          long rowg = rowBase + wave * 16 + r + 8 * (lane >> 4);
          int b = (int)(rowg >> 12), n = (int)(rowg & (N_ - 1));
          logits[((long)(b * H_ + h)) * N_ + n] = logitAcc[r];
        }
      }
#pragma unroll
      for (int r = 0; r < 8; ++r) logitAcc[r] = 0.f;
    }
    wait_async_lds();
    __syncthreads();
  }
}

// ================================================================
// Kernel 2: masked softmax over N, gamma=2 renorm; att in (H,B,N)
// ================================================================
__device__ __forceinline__ float block_reduce(float v, float* red, int t) {
  v += __shfl_xor(v, 1); v += __shfl_xor(v, 2); v += __shfl_xor(v, 4);
  v += __shfl_xor(v, 8); v += __shfl_xor(v, 16);
  __syncthreads();
  if ((t & 31) == 0) red[t >> 5] = v;
  __syncthreads();
  return red[0] + red[1] + red[2] + red[3] + red[4] + red[5] + red[6] + red[7];
}

__global__ __launch_bounds__(256) void k_soft(const float* __restrict__ logits,
                                              const float* __restrict__ masks,
                                              float* __restrict__ att) {
  __shared__ float red[8];
  const int bh = blockIdx.x;                 // b*H + h
  const int b = bh >> 2, h = bh & 3;
  const float* lg = logits + (long)bh * N_;
  const float* mk = masks + (long)b * N_;
  const int t = threadIdx.x;
  float e[16], s = 0.f;
#pragma unroll
  for (int i = 0; i < 16; ++i) {
    int n = t + i * 256;
    float m = mk[n];
    float val = m * __expf(m * lg[n]);
    e[i] = val; s += val;
  }
  s = block_reduce(s, red, t);
  float inv = 1.0f / s, s2 = 0.f;
#pragma unroll
  for (int i = 0; i < 16; ++i) { float a = e[i] * inv; a *= a; e[i] = a; s2 += a; }
  s2 = block_reduce(s2, red, t);
  float inv2 = 1.0f / s2;
  float* out = att + ((long)h * B_ + b) * N_;
#pragma unroll
  for (int i = 0; i < 16; ++i) out[t + i * 256] = e[i] * inv2;
}

// ================================================================
// Kernel 3: emb partials  (split-K over N, deterministic)
// ================================================================
__global__ __launch_bounds__(256) void k_emb(const float* __restrict__ x,
                                             const float* __restrict__ att,
                                             float* __restrict__ partial) {
  const int blk = blockIdx.x;               // b*16 + split
  const int b = blk >> 4, sp = blk & 15;
  const int t = threadIdx.x;
  const int h = t >> 6, c4 = t & 63;
  const float4* xp = (const float4*)x + ((long)b * N_ + sp * 256) * 64 + c4;
  const float* ap = att + ((long)h * B_ + b) * N_ + sp * 256;
  float4 acc = make_float4(0.f, 0.f, 0.f, 0.f);
#pragma unroll 4
  for (int n = 0; n < 256; ++n) {
    float a = ap[n];
    float4 xv = xp[(long)n * 64];
    acc.x += a * xv.x; acc.y += a * xv.y; acc.z += a * xv.z; acc.w += a * xv.w;
  }
  float* dst = partial + (long)sp * (B_ * H_ * M_) + b * (H_ * M_) + h * M_ + c4 * 4;
  dst[0] = acc.x; dst[1] = acc.y; dst[2] = acc.z; dst[3] = acc.w;
}

__global__ __launch_bounds__(256) void k_emb2(const float* __restrict__ partial,
                                              float* __restrict__ emb) {
  int idx = blockIdx.x * 256 + threadIdx.x;    // 0..32767
  float s = 0.f;
#pragma unroll
  for (int sp = 0; sp < 16; ++sp) s += partial[(long)sp * (B_ * H_ * M_) + idx];
  emb[idx] = s;
}

// ================================================================
extern "C" void kernel_launch(void* const* d_in, const int* in_sizes, int n_in,
                              void* d_out, int out_size, void* d_ws, size_t ws_size,
                              hipStream_t stream) {
  (void)in_sizes; (void)n_in; (void)out_size; (void)ws_size;
  const float* x     = (const float*)d_in[0];
  const float* masks = (const float*)d_in[1];
  const float* V     = (const float*)d_in[2];
  const float* U     = (const float*)d_in[3];
  const float* w     = (const float*)d_in[4];

  float* att = (float*)d_out;                          // H*B*N = 524288
  float* emb = (float*)d_out + (long)H_ * B_ * N_;     // B*H*M = 32768

  uint16_t* wfrag  = (uint16_t*)d_ws;                              // 512KB bf16 frags
  float* logits    = (float*)((char*)d_ws + 524288);               // 2MB
  float* partial   = (float*)((char*)d_ws + 524288 + 2097152);     // 2MB

  k_pack<<<1024, 256, 0, stream>>>(V, U, wfrag);
  k_gemm<<<(B_ * N_) / 128, 256, 196608, stream>>>(x, wfrag, w, logits);
  k_soft<<<B_ * H_, 256, 0, stream>>>(logits, masks, att);
  k_emb<<<B_ * 16, 256, 0, stream>>>(x, att, partial);
  k_emb2<<<(B_ * H_ * M_) / 256, 256, 0, stream>>>(partial, emb);
}
